// rits_gru_18897856102808
// MI455X (gfx1250) — matrix operand order repaired and checked by reading, not compiled, not run on hardware
//
#include <hip/hip_runtime.h>

#define V   59
#define VP  64
#define H   128
#define BB  2048
#define TT  96
#define WPB 2
#define BROWS (WPB * 16)

typedef __attribute__((ext_vector_type(16))) _Float16 v16h;
typedef __attribute__((ext_vector_type(8)))  _Float16 v8h;
typedef __attribute__((ext_vector_type(8)))  float    v8f;

// ---- LDS layout (offsets in _Float16 units) ----
// Fragment-major f16 weights (B-operand layout: block=(kt,nt), 32 lanes x 16 halfs)
#define FR_WDH 0           //  2 kt x  8 nt = 16 blocks ( d[16,64]  -> [16,128] )
#define FR_WH  8192        //  4 kt x  4 nt             ( h[16,128] -> [16,64]  )
#define FR_WF  16384       //  2 kt x  4 nt             ( xr[16,64] -> [16,64]  )
#define FR_WC  20480       //  4 kt x  4 nt             ( cat[16,128]-> [16,64] )
#define FR_WIH 28672       //  4 kt x 24 nt             ( cat[16,128]-> [16,384])
#define FR_WHH 77824       //  4 kt x 24 nt             ( h[16,128] -> [16,384] )
#define BIAS_HALF 126976   // 1216 floats of biases / diag
#define STG_HALF  129408   // per-wave staging
#define STG_PER_WAVE 6144  // dbuf 1024 + hbuf 2048 + cbuf 2048 + rbuf 1024
#define SMEM_BYTES ((STG_HALF + WPB * STG_PER_WAVE) * 2)   // 283,392 B < 320 KB

#define LOG2E 1.4426950408889634f

__device__ __forceinline__ v8f wmma_f16(v16h a, v16h b, v8f c) {
  // D = A(16x32 f16) x B(32x16 f16) + C(16x16 f32)
  return __builtin_amdgcn_wmma_f32_16x16x32_f16(false, a, false, b, (short)0, c,
                                                false, false);
}

// A operand: row-major f16 LDS tile, stride halfs; ISA 16-bit A 16x32 layout:
// lanes 0-15: row=lane, K = kt*32 + {0..7, 16..23}; lanes 16-31: K += 8.
__device__ __forceinline__ v16h lds_loadA(const _Float16* buf, int stride, int kt,
                                          int lane) {
  int row = lane & 15;
  int kb  = kt * 32 + ((lane & 16) ? 8 : 0);
  const v8h* p = (const v8h*)(buf + row * stride + kb);
  union { v16h v; v8h h[2]; } u;
  u.h[0] = p[0];     // K = kb .. kb+7
  u.h[1] = p[2];     // K = kb+16 .. kb+23
  return u.v;
}

// B operand: pre-swizzled fragment-major weights, one 32B contiguous read/lane.
__device__ __forceinline__ v16h lds_loadB(const _Float16* frag, int NT, int kt,
                                          int nt, int lane) {
  return *(const v16h*)(frag + (((kt * NT + nt) * 32 + lane) << 4));
}

// Fast transcendentals on the raw HW ops (v_exp_f32 = exp2, v_rcp_f32).
// Range-safe here: exp2(+inf)->inf -> rcp->0 gives the correct saturating limit.
__device__ __forceinline__ float fexp2(float x) { return __builtin_amdgcn_exp2f(x); }
__device__ __forceinline__ float frcp(float x)  { return __builtin_amdgcn_rcpf(x); }
__device__ __forceinline__ float dexp(float x) {           // exp(-relu(x))
  return fexp2(-LOG2E * fmaxf(x, 0.f));
}
__device__ __forceinline__ float sigm(float x) {
  return frcp(1.f + fexp2(-LOG2E * x));
}
__device__ __forceinline__ float ftanh(float x) {          // 2*sigmoid(2x)-1
  return 2.f * frcp(1.f + fexp2(-2.f * LOG2E * x)) - 1.f;
}

// K remap for the [a(59) | pad5 | b(59) | pad5] concat packing into K=128.
__device__ __forceinline__ int kmap_cat(int k) {
  if (k < 59) return k;
  if (k >= 64 && k < 123) return k - 5;
  return -1;
}

// Fill fragment-major f16 B-matrix from row-major f32 W[N][Kw]; B[k][n]=W[n][k].
__device__ void fill_frag(_Float16* dst, const float* W, int KT, int NT, int Neff,
                          int Kw, int kmode, int keff, int zdiag, int tid, int nthr) {
  int total = KT * NT * 512;
  for (int i = tid; i < total; i += nthr) {
    int blk = i >> 9, rem = i & 511;
    int ln = rem >> 4, hh = rem & 15;
    int kt = blk / NT, nt = blk - kt * NT;
    int n = nt * 16 + (ln & 15);
    int k = kt * 32 + ((ln & 16) ? 8 : 0) + (hh & 7) + ((hh & 8) << 1);   // B 32x16: K={0..7,16..23}, lanes16-31 K+8
    int ks = (kmode == 0) ? (k < keff ? k : -1) : kmap_cat(k);
    float v = 0.f;
    if (ks >= 0 && n < Neff) {
      v = W[(size_t)n * Kw + ks];
      if (zdiag && ks == n) v = 0.f;               // Wf zero-diagonal baked in
    }
    dst[i] = (_Float16)v;
  }
}

__global__ __launch_bounds__(WPB * 32, 1)
void rits_main(const float* __restrict__ x, const float* __restrict__ mask,
               const float* __restrict__ deltas,
               const float* __restrict__ Wdh, const float* __restrict__ bdh,
               const float* __restrict__ Wdx, const float* __restrict__ bdx,
               const float* __restrict__ Wh,  const float* __restrict__ bh,
               const float* __restrict__ Wf,  const float* __restrict__ bf,
               const float* __restrict__ Wc,  const float* __restrict__ bc,
               const float* __restrict__ Wih, const float* __restrict__ Whh,
               const float* __restrict__ bih, const float* __restrict__ bhh,
               float* __restrict__ out_imp, float* __restrict__ out_hid,
               float* __restrict__ ws_num, float* __restrict__ ws_den) {
  extern __shared__ _Float16 smem[];
  const int tid = threadIdx.x, nthr = blockDim.x;

  // ---- one-time weight swizzle into LDS (f16, B-fragment-major) ----
  fill_frag(smem + FR_WDH, Wdh, 2,  8, H,     V,     0, V, 0, tid, nthr);
  fill_frag(smem + FR_WH,  Wh,  4,  4, V,     H,     0, H, 0, tid, nthr);
  fill_frag(smem + FR_WF,  Wf,  2,  4, V,     V,     0, V, 1, tid, nthr);
  fill_frag(smem + FR_WC,  Wc,  4,  4, V,     2 * V, 1, 0, 0, tid, nthr);
  fill_frag(smem + FR_WIH, Wih, 4, 24, 3 * H, 2 * V, 1, 0, 0, tid, nthr);
  fill_frag(smem + FR_WHH, Whh, 4, 24, 3 * H, H,     0, H, 0, tid, nthr);

  float* bm = (float*)(smem + BIAS_HALF);
  for (int i = tid; i < H; i += nthr) bm[i] = bdh[i];
  for (int i = tid; i < VP; i += nthr) {
    bm[128 + i] = (i < V) ? bh[i]  : 0.f;
    bm[192 + i] = (i < V) ? bf[i]  : 0.f;
    bm[256 + i] = (i < V) ? bdx[i] : 0.f;
    bm[320 + i] = (i < V) ? Wdx[i * V + i] : 0.f;   // diagonal decay weights
    bm[384 + i] = (i < V) ? bc[i]  : 0.f;
  }
  for (int i = tid; i < 3 * H; i += nthr) { bm[448 + i] = bih[i]; bm[832 + i] = bhh[i]; }
  __syncthreads();

  const float* bdh_s = bm;        const float* bh_s  = bm + 128;
  const float* bf_s  = bm + 192;  const float* bdx_s = bm + 256;
  const float* wdxd  = bm + 320;  const float* bc_s  = bm + 384;
  const float* bih_s = bm + 448;  const float* bhh_s = bm + 832;

  const int wave = tid >> 5, lane = tid & 31;
  const int nl = lane & 15;                 // C/D layout: column within tile
  const int mh = (lane & 16) ? 8 : 0;       // C/D layout: row offset
  const int b0 = blockIdx.x * BROWS + wave * 16;

  _Float16* stg  = smem + STG_HALF + wave * STG_PER_WAVE;
  _Float16* dbuf = stg;            // 16x64  deltas (A staging)
  _Float16* hbuf = stg + 1024;     // 16x128 decayed h (A staging)
  _Float16* cbuf = stg + 3072;     // 16x128 [gamma_x|m] then [x_imp|m]
  _Float16* rbuf = stg + 5120;     // 16x64  x_r

  v8f h[8];
  #pragma unroll
  for (int j = 0; j < 8; ++j) {
    #pragma unroll
    for (int r = 0; r < 8; ++r) h[j][r] = 0.f;
  }

  #pragma unroll 1
  for (int t = 0; t < TT; ++t) {
    // ---- load x, m, d tiles (C/D layout, non-temporal), pad cols >= V with 0 ----
    v8f xt[4], mt[4], dt[4];
    #pragma unroll
    for (int j = 0; j < 4; ++j) {
      int col = j * 16 + nl;
      bool cv = col < V;
      #pragma unroll
      for (int r = 0; r < 8; ++r) {
        size_t base = ((size_t)(b0 + mh + r) * TT + t) * V;
        xt[j][r] = cv ? __builtin_nontemporal_load(x + base + col)      : 0.f;
        mt[j][r] = cv ? __builtin_nontemporal_load(mask + base + col)   : 0.f;
        dt[j][r] = cv ? __builtin_nontemporal_load(deltas + base + col) : 0.f;
      }
    }
    if (t + 1 < TT) {   // global_prefetch_b8: cover next step's 16 rows x 236 B
      #pragma unroll
      for (int r = 0; r < 8; ++r) {
        size_t nb = ((size_t)(b0 + mh + r) * TT + (t + 1)) * V + (nl << 2);
        __builtin_prefetch(&x[nb], 0, 1);
        __builtin_prefetch(&mask[nb], 0, 1);
        __builtin_prefetch(&deltas[nb], 0, 1);
      }
    }

    // ---- stage deltas (f16) as WMMA A ----
    #pragma unroll
    for (int j = 0; j < 4; ++j) {
      #pragma unroll
      for (int r = 0; r < 8; ++r)
        dbuf[(mh + r) * 64 + j * 16 + nl] = (_Float16)dt[j][r];
    }

    // ---- gamma_h = exp(-relu(d @ WdhT + bdh)); h *= gamma_h; stage h ----
    v16h da0 = lds_loadA(dbuf, 64, 0, lane);
    v16h da1 = lds_loadA(dbuf, 64, 1, lane);
    #pragma unroll
    for (int nt = 0; nt < 8; ++nt) {
      v8f acc = {};
      acc = wmma_f16(da0, lds_loadB(smem + FR_WDH, 8, 0, nt, lane), acc);
      acc = wmma_f16(da1, lds_loadB(smem + FR_WDH, 8, 1, nt, lane), acc);
      float bv = bdh_s[nt * 16 + nl];
      #pragma unroll
      for (int r = 0; r < 8; ++r) {
        float g = dexp(acc[r] + bv);
        h[nt][r] *= g;
        hbuf[(mh + r) * 128 + nt * 16 + nl] = (_Float16)h[nt][r];
      }
    }

    // ---- x_h = h @ WhT + bh (ha fragments reused later by the GRU) ----
    v16h ha[4];
    #pragma unroll
    for (int kt = 0; kt < 4; ++kt) ha[kt] = lds_loadA(hbuf, 128, kt, lane);
    v8f xh[4];
    #pragma unroll
    for (int nt = 0; nt < 4; ++nt) {
      v8f acc = {};
      #pragma unroll
      for (int kt = 0; kt < 4; ++kt)
        acc = wmma_f16(ha[kt], lds_loadB(smem + FR_WH, 4, kt, nt, lane), acc);
      float bv = bh_s[nt * 16 + nl];
      #pragma unroll
      for (int r = 0; r < 8; ++r) acc[r] += bv;
      xh[nt] = acc;
    }

    // ---- x_r = m*x + (1-m)*x_h ; stage f16 ----
    #pragma unroll
    for (int j = 0; j < 4; ++j) {
      #pragma unroll
      for (int r = 0; r < 8; ++r) {
        float v = mt[j][r] * xt[j][r] + (1.f - mt[j][r]) * xh[j][r];
        rbuf[(mh + r) * 64 + j * 16 + nl] = (_Float16)v;
      }
    }

    // ---- xu = x_r @ Wf(0-diag)T + bf ----
    v16h ra0 = lds_loadA(rbuf, 64, 0, lane);
    v16h ra1 = lds_loadA(rbuf, 64, 1, lane);
    v8f xu[4];
    #pragma unroll
    for (int nt = 0; nt < 4; ++nt) {
      v8f acc = {};
      acc = wmma_f16(ra0, lds_loadB(smem + FR_WF, 4, 0, nt, lane), acc);
      acc = wmma_f16(ra1, lds_loadB(smem + FR_WF, 4, 1, nt, lane), acc);
      float bv = bf_s[nt * 16 + nl];
      #pragma unroll
      for (int r = 0; r < 8; ++r) acc[r] += bv;
      xu[nt] = acc;
    }

    // ---- gamma_x (diagonal decay) ; stage [gamma_x | m] ----
    #pragma unroll
    for (int j = 0; j < 4; ++j) {
      int col = j * 16 + nl;
      float wv = wdxd[col], bv = bdx_s[col];
      #pragma unroll
      for (int r = 0; r < 8; ++r) {
        float g = dexp(dt[j][r] * wv + bv);
        cbuf[(mh + r) * 128 + col]      = (_Float16)g;
        cbuf[(mh + r) * 128 + 64 + col] = (_Float16)mt[j][r];
      }
    }

    // ---- beta = [gamma_x|m] @ WcT + bc ; x_comb ; loss ; x_imp ----
    v16h ca[4];
    #pragma unroll
    for (int kt = 0; kt < 4; ++kt) ca[kt] = lds_loadA(cbuf, 128, kt, lane);
    float lnum = 0.f, lden = 0.f;
    #pragma unroll
    for (int nt = 0; nt < 4; ++nt) {
      v8f acc = {};
      #pragma unroll
      for (int kt = 0; kt < 4; ++kt)
        acc = wmma_f16(ca[kt], lds_loadB(smem + FR_WC, 4, kt, nt, lane), acc);
      float bv = bc_s[nt * 16 + nl];
      int col = nt * 16 + nl;
      #pragma unroll
      for (int r = 0; r < 8; ++r) {
        float beta = acc[r] + bv;
        float xc = beta * xu[nt][r] + (1.f - beta) * xh[nt][r];
        float m  = mt[nt][r];
        lnum += fabsf(xt[nt][r] - xc) * m;
        lden += m;
        float xi = m * xt[nt][r] + (1.f - m) * xc;
        cbuf[(mh + r) * 128 + col] = (_Float16)xi;   // overwrite gamma_x -> x_imp
        if (col < V)
          __builtin_nontemporal_store(
              xi, out_imp + ((size_t)(b0 + mh + r) * TT + t) * V + col);
      }
    }
    #pragma unroll
    for (int off = 16; off > 0; off >>= 1) {
      lnum += __shfl_xor(lnum, off, 32);
      lden += __shfl_xor(lden, off, 32);
    }
    if (lane == 0) {
      atomicAdd(&ws_num[t], lnum);
      atomicAdd(&ws_den[t], lden);
    }

    // ---- GRU: gi = [x_imp|m] @ WihT + bih ; gh = h @ WhhT + bhh ----
    v16h cf[4];
    #pragma unroll
    for (int kt = 0; kt < 4; ++kt) cf[kt] = lds_loadA(cbuf, 128, kt, lane);
    #pragma unroll
    for (int jj = 0; jj < 8; ++jj) {
      v8f ir = {}, iz = {}, in_ = {}, hr = {}, hz = {}, hn = {};
      #pragma unroll
      for (int kt = 0; kt < 4; ++kt) {
        ir  = wmma_f16(cf[kt], lds_loadB(smem + FR_WIH, 24, kt, jj,      lane), ir);
        iz  = wmma_f16(cf[kt], lds_loadB(smem + FR_WIH, 24, kt, 8 + jj,  lane), iz);
        in_ = wmma_f16(cf[kt], lds_loadB(smem + FR_WIH, 24, kt, 16 + jj, lane), in_);
        hr  = wmma_f16(ha[kt], lds_loadB(smem + FR_WHH, 24, kt, jj,      lane), hr);
        hz  = wmma_f16(ha[kt], lds_loadB(smem + FR_WHH, 24, kt, 8 + jj,  lane), hz);
        hn  = wmma_f16(ha[kt], lds_loadB(smem + FR_WHH, 24, kt, 16 + jj, lane), hn);
      }
      int c = jj * 16 + nl;
      float bir = bih_s[c], biz = bih_s[128 + c], bin = bih_s[256 + c];
      float bhr = bhh_s[c], bhz = bhh_s[128 + c], bhn = bhh_s[256 + c];
      #pragma unroll
      for (int r = 0; r < 8; ++r) {
        float rg = sigm(ir[r] + bir + hr[r] + bhr);
        float zg = sigm(iz[r] + biz + hz[r] + bhz);
        float ng = ftanh(in_[r] + bin + rg * (hn[r] + bhn));
        float hv = (1.f - zg) * ng + zg * h[jj][r];
        h[jj][r] = hv;
        __builtin_nontemporal_store(
            hv, out_hid + ((size_t)(b0 + mh + r) * TT + t) * H + c);
      }
    }
  }
}

__global__ void rits_loss(const float* __restrict__ num, const float* __restrict__ den,
                          float* __restrict__ out) {
  if (threadIdx.x == 0 && blockIdx.x == 0) {
    float s = 0.f;
    for (int t = 0; t < TT; ++t) s += num[t] / (den[t] + 1e-5f);
    out[0] = s;
  }
}

extern "C" void kernel_launch(void* const* d_in, const int* in_sizes, int n_in,
                              void* d_out, int out_size, void* d_ws, size_t ws_size,
                              hipStream_t stream) {
  (void)in_sizes; (void)n_in; (void)out_size; (void)ws_size;
  const float* x      = (const float*)d_in[0];
  const float* mask   = (const float*)d_in[1];
  const float* deltas = (const float*)d_in[2];
  const float* Wdh    = (const float*)d_in[3];
  const float* bdh    = (const float*)d_in[4];
  const float* Wdx    = (const float*)d_in[5];
  const float* bdx    = (const float*)d_in[6];
  const float* Wh     = (const float*)d_in[7];
  const float* bh     = (const float*)d_in[8];
  const float* Wf     = (const float*)d_in[9];
  const float* bf     = (const float*)d_in[10];
  const float* Wc     = (const float*)d_in[11];
  const float* bc     = (const float*)d_in[12];
  const float* Wih    = (const float*)d_in[13];
  const float* Whh    = (const float*)d_in[14];
  const float* bih    = (const float*)d_in[15];
  const float* bhh    = (const float*)d_in[16];

  float* out      = (float*)d_out;
  float* out_imp  = out;                                  // [B,T,V]
  float* out_loss = out + (size_t)BB * TT * V;            // scalar
  float* out_hid  = out_loss + 1;                         // [B,T,H]
  float* ws_num = (float*)d_ws;
  float* ws_den = ws_num + TT;

  hipMemsetAsync(d_ws, 0, 2 * TT * sizeof(float), stream);
  rits_main<<<BB / BROWS, WPB * 32, SMEM_BYTES, stream>>>(
      x, mask, deltas, Wdh, bdh, Wdx, bdx, Wh, bh, Wf, bf, Wc, bc,
      Wih, Whh, bih, bhh, out_imp, out_hid, ws_num, ws_den);
  rits_loss<<<1, 32, 0, stream>>>(ws_num, ws_den, out_loss);
}
